// CorrectError_91199335563348
// MI455X (gfx1250) — matrix operand order repaired and checked by reading, not compiled, not run on hardware
//
#include <hip/hip_runtime.h>
#include <hip/hip_bf16.h>
#include <stdint.h>

#define D_DIM 1024

typedef __attribute__((ext_vector_type(16))) __bf16 v16bf;
typedef __attribute__((ext_vector_type(8)))  float  v8f;
typedef __attribute__((ext_vector_type(4)))  int    v4i;

struct Cand { float d; int i; };

union FragBF {
    v16bf v;
    uint4 q[2];
};

__device__ __forceinline__ void ins3(float d, int i,
                                     float& d0, float& d1, float& d2,
                                     int& i0, int& i1, int& i2) {
    if (d < d2) {
        if (d < d0)      { d2 = d1; i2 = i1; d1 = d0; i1 = i0; d0 = d; i0 = i; }
        else if (d < d1) { d2 = d1; i2 = i1; d1 = d;  i1 = i; }
        else             { d2 = d;  i2 = i; }
    }
}

#if defined(__gfx1250__) && __has_builtin(__builtin_amdgcn_global_load_async_to_lds_b128)
#define HAVE_ASYNC_LDS 1
#else
#define HAVE_ASYNC_LDS 0
#endif

// -------- phase 0: fp32 -> bf16 conversion fused with row ||.||^2 --------
__global__ __launch_bounds__(256) void k_convert(const float* __restrict__ src,
                                                 __bf16* __restrict__ dst,
                                                 float* __restrict__ sq,
                                                 int nrows) {
    const int row = blockIdx.x;
    if (row >= nrows) return;
    const float4* s = (const float4*)(src + (size_t)row * D_DIM);
    __bf16* d = dst + (size_t)row * D_DIM;

    const int i = threadIdx.x;              // 256 threads * 4 elems = 1024
    float4 v = s[i];
    float acc = v.x * v.x + v.y * v.y + v.z * v.z + v.w * v.w;
    __bf16 h[4] = { (__bf16)v.x, (__bf16)v.y, (__bf16)v.z, (__bf16)v.w };
    __builtin_memcpy(d + i * 4, h, 8);

    #pragma unroll
    for (int off = 16; off > 0; off >>= 1) acc += __shfl_down(acc, off);
    __shared__ float part[8];
    if ((threadIdx.x & 31) == 0) part[threadIdx.x >> 5] = acc;
    __syncthreads();
    if (threadIdx.x == 0) {
        float t = 0.f;
        #pragma unroll
        for (int w = 0; w < 8; ++w) t += part[w];
        sq[row] = t;
    }
}

__global__ void k_zero(float* sum) { *sum = 0.f; }

// -------- phase 1: bf16 WMMA GEMM + per-(query, col-block) top-3 --------
__global__ __launch_bounds__(128) void k_gemm_topk(
    const __bf16* __restrict__ qb, const __bf16* __restrict__ mb,
    const float* __restrict__ qsq, const float* __restrict__ msq,
    Cand* __restrict__ cand, int N, int numCB)
{
    constexpr int LDA = D_DIM + 8;                     // pad 16B: bank-conflict-free
    __shared__ __align__(16) __bf16 As[16 * LDA];      // 16 x 1024 A tile (33 KB)
    __shared__ float cd[4][16][16][3];
    __shared__ int   ci[4][16][16][3];

    const int row_tile = blockIdx.y;
    const int cb  = blockIdx.x;
    const int col0 = cb * 256;
    const int tid  = threadIdx.x;
    const int wave = tid >> 5;
    const int lane = tid & 31;
    const int hi   = (lane & 16) ? 1 : 0;              // lane group (K-half select)
    const int l15  = lane & 15;

    // stage A tile: 16 rows x 1024 K, 8 threads per row, 128 contiguous bf16 each
    {
        const int r = tid >> 3;
        const int cpart = (tid & 7) * 128;
        const __bf16* srcp = qb + ((size_t)row_tile * 16 + r) * D_DIM + cpart;
        __bf16* dstp = As + r * LDA + cpart;
#if HAVE_ASYNC_LDS
        // direct memory -> LDS, bypassing VGPRs; tracked with ASYNCcnt
        auto gsrc = (__attribute__((address_space(1))) v4i*)(srcp);
        auto ldst = (__attribute__((address_space(3))) v4i*)(dstp);
        #pragma unroll
        for (int u = 0; u < 16; ++u)
            __builtin_amdgcn_global_load_async_to_lds_b128(gsrc + u, ldst + u, 0, 0);
#if __has_builtin(__builtin_amdgcn_s_wait_asynccnt)
        __builtin_amdgcn_s_wait_asynccnt(0);
#else
        asm volatile("s_wait_asynccnt 0x0" ::: "memory");
#endif
#else
        const uint4* s4 = (const uint4*)srcp;
        uint4* d4 = (uint4*)dstp;
        #pragma unroll
        for (int u = 0; u < 16; ++u) d4[u] = s4[u];
#endif
    }
    __syncthreads();

    float qs[8];
    {
        const int m0 = row_tile * 16 + hi * 8;         // C layout: lanes>=16 hold M=j+8
        #pragma unroll
        for (int j = 0; j < 8; ++j) qs[j] = qsq[m0 + j];
    }

    float bd[8][3];
    int   bi[8][3];
    #pragma unroll
    for (int j = 0; j < 8; ++j) {
        #pragma unroll
        for (int s = 0; s < 3; ++s) { bd[j][s] = 3.0e38f; bi[j][s] = 0; }
    }

    // 16 column tiles per block, 4 per wave, processed as 2 concurrent pairs:
    // two independent WMMA accumulator chains sharing one A fragment per K step.
    for (int t = 0; t < 2; ++t) {
        const int ct0 = wave * 4 + t * 2;
        const int n0 = col0 + ct0 * 16 + l15;
        const int n1 = n0 + 16;
        const int nc0 = (n0 < N) ? n0 : (N - 1);
        const int nc1 = (n1 < N) ? n1 : (N - 1);
        const __bf16* brow0 = mb + (size_t)nc0 * D_DIM + hi * 16;
        const __bf16* brow1 = mb + (size_t)nc1 * D_DIM + hi * 16;

        v8f c0 = {};
        v8f c1 = {};
        for (int k = 0; k < D_DIM; k += 32) {
            __builtin_prefetch(brow0 + k + 64, 0, 0);  // stream next K chunk
            __builtin_prefetch(brow1 + k + 64, 0, 0);
            FragBF a, b0, b1;
            // A frag: lanes<16 K={k..k+15}; lanes>=16 K={k+16..k+31}
            const __bf16* ap = As + (size_t)l15 * LDA + k + hi * 16;
            a.q[0] = *(const uint4*)(ap);
            a.q[1] = *(const uint4*)(ap + 8);
            // B frags: 16 contiguous bf16 of memory row (K half per lane group)
            b0.q[0] = *(const uint4*)(brow0 + k);
            b0.q[1] = *(const uint4*)(brow0 + k + 8);
            b1.q[0] = *(const uint4*)(brow1 + k);
            b1.q[1] = *(const uint4*)(brow1 + k + 8);
            c0 = __builtin_amdgcn_wmma_f32_16x16x32_bf16(false, a.v, false, b0.v,
                                                         (short)0, c0, false, false);
            c1 = __builtin_amdgcn_wmma_f32_16x16x32_bf16(false, a.v, false, b1.v,
                                                         (short)0, c1, false, false);
        }

        if (n0 < N) {
            const float ms = msq[nc0];
            #pragma unroll
            for (int j = 0; j < 8; ++j) {
                const float dist = qs[j] + ms - 2.0f * c0[j];
                ins3(dist, n0, bd[j][0], bd[j][1], bd[j][2],
                               bi[j][0], bi[j][1], bi[j][2]);
            }
        }
        if (n1 < N) {
            const float ms = msq[nc1];
            #pragma unroll
            for (int j = 0; j < 8; ++j) {
                const float dist = qs[j] + ms - 2.0f * c1[j];
                ins3(dist, n1, bd[j][0], bd[j][1], bd[j][2],
                               bi[j][0], bi[j][1], bi[j][2]);
            }
        }
    }

    // cross-lane / cross-wave merge via LDS
    #pragma unroll
    for (int j = 0; j < 8; ++j) {
        #pragma unroll
        for (int s = 0; s < 3; ++s) {
            cd[wave][hi * 8 + j][l15][s] = bd[j][s];
            ci[wave][hi * 8 + j][l15][s] = bi[j][s];
        }
    }
    __syncthreads();

    if (tid < 16) {
        float d0 = 3.0e38f, d1 = 3.0e38f, d2 = 3.0e38f;
        int i0 = 0, i1 = 0, i2 = 0;
        for (int w = 0; w < 4; ++w)
            for (int l = 0; l < 16; ++l)
                for (int s = 0; s < 3; ++s)
                    ins3(cd[w][tid][l][s], ci[w][tid][l][s], d0, d1, d2, i0, i1, i2);
        const size_t base = ((size_t)(row_tile * 16 + tid) * numCB + cb) * 3;
        cand[base + 0] = { d0, i0 };
        cand[base + 1] = { d1, i1 };
        cand[base + 2] = { d2, i2 };
    }
}

// -------- phase 2: per-query candidate merge + gather + global sum --------
__global__ __launch_bounds__(256) void k_merge(const Cand* __restrict__ cand,
                                               const float* __restrict__ pv,
                                               float* __restrict__ sum,
                                               int numCB, int B) {
    const int q = blockIdx.x * blockDim.x + threadIdx.x;
    if (q >= B) return;
    float d0 = 3.0e38f, d1 = 3.0e38f, d2 = 3.0e38f;
    int i0 = 0, i1 = 0, i2 = 0;
    const Cand* c = cand + (size_t)q * numCB * 3;
    const int total = numCB * 3;
    for (int t = 0; t < total; ++t)
        ins3(c[t].d, c[t].i, d0, d1, d2, i0, i1, i2);
    const float s = pv[i0] + pv[i1] + pv[i2];
    atomicAdd(sum, s);
}

__global__ void k_finalize(const float* sum, float* out, float denom) {
    out[0] = sum[0] / denom;
}

// ------------------------------ host glue ------------------------------
extern "C" void kernel_launch(void* const* d_in, const int* in_sizes, int n_in,
                              void* d_out, int out_size, void* d_ws, size_t ws_size,
                              hipStream_t stream) {
    const float* hq = (const float*)d_in[0];   // [B, D]
    const float* me = (const float*)d_in[1];   // [N, D]
    const float* pv = (const float*)d_in[2];   // [N]
    float* out = (float*)d_out;

    const int B = in_sizes[0] / D_DIM;
    const int N = in_sizes[2];
    const int numCB = (N + 255) / 256;
    const int rowTiles = B / 16;

    char* ws = (char*)d_ws;
    auto align256 = [](size_t x) { return (x + 255) & ~(size_t)255; };
    size_t off = 0;
    __bf16* mbf = (__bf16*)(ws + off); off = align256(off + (size_t)N * D_DIM * 2);
    __bf16* qbf = (__bf16*)(ws + off); off = align256(off + (size_t)B * D_DIM * 2);
    float* msq  = (float*)(ws + off);  off = align256(off + (size_t)N * 4);
    float* qsq  = (float*)(ws + off);  off = align256(off + (size_t)B * 4);
    Cand* cand  = (Cand*)(ws + off);   off = align256(off + (size_t)B * numCB * 3 * sizeof(Cand));
    float* sum  = (float*)(ws + off);  off += 256;

    k_convert<<<N, 256, 0, stream>>>(me, mbf, msq, N);
    k_convert<<<B, 256, 0, stream>>>(hq, qbf, qsq, B);
    k_zero<<<1, 1, 0, stream>>>(sum);
    k_gemm_topk<<<dim3(numCB, rowTiles), 128, 0, stream>>>(qbf, mbf, qsq, msq,
                                                           cand, N, numCB);
    k_merge<<<(B + 255) / 256, 256, 0, stream>>>(cand, pv, sum, numCB, B);
    k_finalize<<<1, 1, 0, stream>>>(sum, out, (float)(B * 3));
}
